// MultiheadAttention_36936718745879
// MI455X (gfx1250) — matrix operand order repaired and checked by reading, not compiled, not run on hardware
//
#include <hip/hip_runtime.h>
#include <hip/hip_bf16.h>

#define Bq  2
#define Sq  4096
#define Dq  768
#define Hq  12
#define HDq 64
#define LDT 40   // LDS row stride in bf16 elems (32 + 8 pad) -> 80B, conflict-free

typedef __attribute__((ext_vector_type(16))) __bf16 v16bf;
typedef __attribute__((ext_vector_type(8)))  __bf16 v8bf;
typedef __attribute__((ext_vector_type(4)))  __bf16 v4bf;
typedef __attribute__((ext_vector_type(8)))  float  v8f;
typedef __attribute__((ext_vector_type(4)))  int    v4i;

#if defined(__has_builtin)
#if __has_builtin(__builtin_amdgcn_global_load_async_to_lds_b128) && \
    __has_builtin(__builtin_amdgcn_s_wait_asynccnt)
#define USE_ASYNC_LDS 1
#endif
#endif
#ifndef USE_ASYNC_LDS
#define USE_ASYNC_LDS 0
#endif

static __device__ __forceinline__ v16bf cat16(v8bf lo, v8bf hi) {
  v16bf r;
#pragma unroll
  for (int i = 0; i < 8; ++i) { r[i] = lo[i]; r[i + 8] = hi[i]; }
  return r;
}

static __device__ __forceinline__ v8f wmma_bf16(v16bf a, v16bf b, v8f c) {
  // D(16x16,f32) = A(16x32,bf16) x B(32x16,bf16) + C
  return __builtin_amdgcn_wmma_f32_16x16x32_bf16(false, a, false, b, (short)0, c,
                                                 false, false);
}

// 16-byte global -> LDS copy; async path uses the CDNA5 ASYNCcnt-tracked DMA.
// Builtin signature (from toolchain diagnostic): (v4i as1*, v4i as3*, imm, imm).
static __device__ __forceinline__ void cp16_g2l(const __bf16* g, __bf16* l) {
#if USE_ASYNC_LDS
  v4i* gp = (v4i*)g;   // reinterpret (drops const), still generic AS
  v4i* lp = (v4i*)l;
  __builtin_amdgcn_global_load_async_to_lds_b128(
      (__attribute__((address_space(1))) v4i*)gp,
      (__attribute__((address_space(3))) v4i*)lp, 0, 0);
#else
  *(v8bf*)l = *(const v8bf*)g;
#endif
}

static __device__ __forceinline__ void cp16_fence_then_barrier() {
#if USE_ASYNC_LDS
  __builtin_amdgcn_s_wait_asynccnt(0);
#endif
  __syncthreads();
}

// ---------------------------------------------------------------------------
// Kernel 1: fused QKV projection.  out[m,n] = sum_k x[m,k] * W[n,k]  (NT GEMM)
// Q,K written [b,h,s,hd]; V written TRANSPOSED [b,h,hd,s] so the attention
// kernel can stage V tiles with pure contiguous copies (no per-tile transpose).
// grid = (M/64, N/64, 3), block = 128 (4 waves, each 32x32 of the 64x64 tile)
// ---------------------------------------------------------------------------
__global__ void __launch_bounds__(128)
mha_qkv_gemm(const float* __restrict__ x,  const float* __restrict__ wq,
             const float* __restrict__ wk, const float* __restrict__ wv,
             __bf16* __restrict__ qo, __bf16* __restrict__ ko,
             __bf16* __restrict__ vo)
{
  __shared__ __align__(16) __bf16 As[64 * LDT];
  __shared__ __align__(16) __bf16 Bs[64 * LDT];
  const int tid  = threadIdx.x;
  const int lane = tid & 31, wave = tid >> 5;
  const int lh   = lane & 15, kh = lane >> 4;       // half-wave id per ISA layout
  const int wm   = wave >> 1, wn = wave & 1;
  const int m0   = blockIdx.x * 64, n0 = blockIdx.y * 64;
  const float* w    = (blockIdx.z == 0) ? wq : (blockIdx.z == 1 ? wk : wv);
  __bf16*      outp = (blockIdx.z == 0) ? qo : (blockIdx.z == 1 ? ko : vo);

  v8f zf = {};
  v8f acc[2][2] = {{zf, zf}, {zf, zf}};

  for (int k0 = 0; k0 < Dq; k0 += 32) {
#pragma unroll
    for (int p = 0; p < 4; ++p) {                   // 64x32 f32 -> bf16 LDS
      int lin = (p * 128 + tid) * 4;
      int r = lin >> 5, c = lin & 31;
      const float* xg = x + (size_t)(m0 + r) * Dq + k0 + c;
      const float* wg = w + (size_t)(n0 + r) * Dq + k0 + c;
      float4 av = *(const float4*)xg;
      float4 bv = *(const float4*)wg;
      // Speculative prefetch of next K-panel (ISA 10.5: OOB silently dropped),
      // unconditional to keep the hot loop branch-free.
      __builtin_prefetch(xg + 32, 0, 1);
      __builtin_prefetch(wg + 32, 0, 1);
      v4bf a4; a4[0]=(__bf16)av.x; a4[1]=(__bf16)av.y; a4[2]=(__bf16)av.z; a4[3]=(__bf16)av.w;
      v4bf b4; b4[0]=(__bf16)bv.x; b4[1]=(__bf16)bv.y; b4[2]=(__bf16)bv.z; b4[3]=(__bf16)bv.w;
      *(v4bf*)&As[r * LDT + c] = a4;
      *(v4bf*)&Bs[r * LDT + c] = b4;
    }
    __syncthreads();
    v16bf af[2], bfr[2];
#pragma unroll
    for (int mt = 0; mt < 2; ++mt) {                // A layout: K in {kh*8..+7, 16+kh*8..+7}
      const __bf16* p = &As[(wm * 32 + mt * 16 + lh) * LDT + kh * 8];
      af[mt] = cat16(*(const v8bf*)p, *(const v8bf*)(p + 16));
    }
#pragma unroll
    for (int nt = 0; nt < 2; ++nt) {                // B layout: K in {kh*8..+7, 16+kh*8..+7}
      const __bf16* p = &Bs[(wn * 32 + nt * 16 + lh) * LDT + kh * 8];
      bfr[nt] = cat16(*(const v8bf*)p, *(const v8bf*)(p + 16));
    }
#pragma unroll
    for (int mt = 0; mt < 2; ++mt)
#pragma unroll
      for (int nt = 0; nt < 2; ++nt)
        acc[mt][nt] = wmma_bf16(af[mt], bfr[nt], acc[mt][nt]);
    __syncthreads();
  }

#pragma unroll
  for (int mt = 0; mt < 2; ++mt)
#pragma unroll
    for (int nt = 0; nt < 2; ++nt)
#pragma unroll
      for (int r = 0; r < 8; ++r) {                 // C layout: M = r + kh*8, N = lh
        int m = m0 + wm * 32 + mt * 16 + r + kh * 8;
        int n = n0 + wn * 32 + nt * 16 + lh;
        int b = m >> 12, s = m & (Sq - 1);
        int h = n >> 6,  hd = n & 63;
        __bf16 val = (__bf16)acc[mt][nt][r];
        if (blockIdx.z != 2)                        // Q,K: [b,h,s,hd]
          outp[(((size_t)(b * Hq + h)) * Sq + s) * HDq + hd] = val;
        else                                        // V:   [b,h,hd,s]
          outp[(((size_t)(b * Hq + h)) * HDq + hd) * Sq + s] = val;
      }
}

// ---------------------------------------------------------------------------
// Kernel 2: causal flash attention per (b,h).  block = 128 thr = 4 waves,
// each wave owns 32 query rows; key tiles of 32, online softmax in f32.
// K staged [key][d], V^T staged [hd][key] straight from the pre-transposed
// global layout (async global->LDS DMA when available + prefetch of next tile).
// grid = (S/128, B*H)
// ---------------------------------------------------------------------------
__global__ void __launch_bounds__(128)
mha_flash_attn(const __bf16* __restrict__ qb, const __bf16* __restrict__ kb,
               const __bf16* __restrict__ vtb, __bf16* __restrict__ ctx)
{
  __shared__ __align__(16) __bf16 Ks[32 * 72];      // [key][d]
  __shared__ __align__(16) __bf16 Vts[64 * 40];     // [hd][key]
  __shared__ __align__(16) __bf16 Ps[4 * 32 * 40];  // per-wave P staging [row][key]
  const int tid  = threadIdx.x, lane = tid & 31, wave = tid >> 5;
  const int lh   = lane & 15, kh = lane >> 4;
  const int bh   = blockIdx.y;
  const int q0   = blockIdx.x * 128 + wave * 32;
  const __bf16* Q  = qb  + (size_t)bh * Sq * HDq;
  const __bf16* K  = kb  + (size_t)bh * Sq * HDq;
  const __bf16* Vt = vtb + (size_t)bh * Sq * HDq;   // [hd][s]

  v16bf qf[2][2];                                   // [mtile][d-chunk of 32]
#pragma unroll
  for (int mt = 0; mt < 2; ++mt)
#pragma unroll
    for (int kc = 0; kc < 2; ++kc) {
      const __bf16* p = Q + (size_t)(q0 + mt * 16 + lh) * HDq + kc * 32 + kh * 8;
      qf[mt][kc] = cat16(*(const v8bf*)p, *(const v8bf*)(p + 16));
    }

  float mrow[2][8], lrow[2][8];
  v8f zf = {};
  v8f Oacc[2][4];
#pragma unroll
  for (int mt = 0; mt < 2; ++mt) {
#pragma unroll
    for (int r = 0; r < 8; ++r) { mrow[mt][r] = -__builtin_inff(); lrow[mt][r] = 0.f; }
#pragma unroll
    for (int nto = 0; nto < 4; ++nto) Oacc[mt][nto] = zf;
  }

  const int ktiles = blockIdx.x * 4 + 4;            // causal upper bound for block
  for (int kt = 0; kt < ktiles; ++kt) {
    const int k0 = kt * 32;
#pragma unroll
    for (int p = 0; p < 2; ++p) {                   // 256 x 16B chunks each tile
      int c2 = p * 128 + tid;                       // 0..255
      int kr = c2 >> 3, kc8 = (c2 & 7) * 8;         // K tile:  32 rows x 128B
      int vr = c2 >> 2, vc8 = (c2 & 3) * 8;         // Vt tile: 64 rows x 64B
      const __bf16* kg = K  + (size_t)(k0 + kr) * HDq + kc8;
      const __bf16* vg = Vt + (size_t)vr * Sq + k0 + vc8;
      cp16_g2l(kg, &Ks[kr * 72 + kc8]);
      cp16_g2l(vg, &Vts[vr * 40 + vc8]);
      // Unconditional speculative prefetch of next key tile (branch-free).
      __builtin_prefetch(kg + 32 * HDq, 0, 1);
      __builtin_prefetch(vg + 32, 0, 1);
    }
    cp16_fence_then_barrier();

    v16bf kf[2][2];                                 // B frags of K^T: lane = key, K = d
#pragma unroll
    for (int nt = 0; nt < 2; ++nt)
#pragma unroll
      for (int kc = 0; kc < 2; ++kc) {
        const __bf16* p = &Ks[(nt * 16 + lh) * 72 + kc * 32 + kh * 8];
        kf[nt][kc] = cat16(*(const v8bf*)p, *(const v8bf*)(p + 16));
      }

    v8f sc[2][2];
#pragma unroll
    for (int mt = 0; mt < 2; ++mt)
#pragma unroll
      for (int nt = 0; nt < 2; ++nt) {
        v8f t = wmma_bf16(qf[mt][0], kf[nt][0], zf);     // d = 0..31
        sc[mt][nt] = wmma_bf16(qf[mt][1], kf[nt][1], t); // d = 32..63
      }

    __bf16* ps = &Ps[wave * 32 * 40];
#pragma unroll
    for (int mt = 0; mt < 2; ++mt)
#pragma unroll
      for (int r = 0; r < 8; ++r) {
        const int qi = q0 + mt * 16 + r + kh * 8;
        float s0 = sc[mt][0][r] * 0.125f;           // 1/sqrt(64)
        float s1 = sc[mt][1][r] * 0.125f;
        if (k0 + lh > qi)      s0 = -__builtin_inff();
        if (k0 + 16 + lh > qi) s1 = -__builtin_inff();
        float v = fmaxf(s0, s1);                    // row-max across 16 lanes
        v = fmaxf(v, __shfl_xor(v, 1));
        v = fmaxf(v, __shfl_xor(v, 2));
        v = fmaxf(v, __shfl_xor(v, 4));
        v = fmaxf(v, __shfl_xor(v, 8));
        float mold  = mrow[mt][r];
        float mnew  = fmaxf(mold, v);
        float alpha = __expf(mold - mnew);
        mrow[mt][r] = mnew;
        float p0 = __expf(s0 - mnew);
        float p1 = __expf(s1 - mnew);
        float rs = p0 + p1;                         // row-sum
        rs += __shfl_xor(rs, 1);
        rs += __shfl_xor(rs, 2);
        rs += __shfl_xor(rs, 4);
        rs += __shfl_xor(rs, 8);
        lrow[mt][r] = lrow[mt][r] * alpha + rs;
#pragma unroll
        for (int nto = 0; nto < 4; ++nto) Oacc[mt][nto][r] *= alpha;
        const int row = mt * 16 + r + kh * 8;       // C-layout -> LDS [row][key]
        ps[row * 40 + lh]      = (__bf16)p0;
        ps[row * 40 + 16 + lh] = (__bf16)p1;
      }

    v16bf pf[2];                                    // same-wave DS is in-order: safe
#pragma unroll
    for (int mt = 0; mt < 2; ++mt) {
      const __bf16* p = &ps[(mt * 16 + lh) * 40 + kh * 8];
      pf[mt] = cat16(*(const v8bf*)p, *(const v8bf*)(p + 16));
    }
#pragma unroll
    for (int nto = 0; nto < 4; ++nto) {             // B frags of V: lane = hd, K = key
      const __bf16* p = &Vts[(nto * 16 + lh) * 40 + kh * 8];
      v16bf vf = cat16(*(const v8bf*)p, *(const v8bf*)(p + 16));
#pragma unroll
      for (int mt = 0; mt < 2; ++mt)
        Oacc[mt][nto] = wmma_bf16(pf[mt], vf, Oacc[mt][nto]);
    }
    __syncthreads();
  }

  const int b = bh / Hq, h = bh % Hq;
#pragma unroll
  for (int mt = 0; mt < 2; ++mt)
#pragma unroll
    for (int nto = 0; nto < 4; ++nto)
#pragma unroll
      for (int r = 0; r < 8; ++r) {
        int s  = q0 + mt * 16 + r + kh * 8;
        int hd = nto * 16 + lh;
        float o = Oacc[mt][nto][r] / lrow[mt][r];
        ctx[((size_t)(b * Sq + s)) * Dq + h * HDq + hd] = (__bf16)o;
      }
}

// ---------------------------------------------------------------------------
// Kernel 3: output projection.  out[m,n] = sum_k ctx[m,k]*Wo[n,k] + bias[n]
// grid = (M/64, N/64), block = 128
// ---------------------------------------------------------------------------
__global__ void __launch_bounds__(128)
mha_out_gemm(const __bf16* __restrict__ ctxb, const float* __restrict__ wo,
             const float* __restrict__ bo, float* __restrict__ out)
{
  __shared__ __align__(16) __bf16 As[64 * LDT];
  __shared__ __align__(16) __bf16 Bs[64 * LDT];
  const int tid  = threadIdx.x;
  const int lane = tid & 31, wave = tid >> 5;
  const int lh   = lane & 15, kh = lane >> 4;
  const int wm   = wave >> 1, wn = wave & 1;
  const int m0   = blockIdx.x * 64, n0 = blockIdx.y * 64;

  v8f zf = {};
  v8f acc[2][2] = {{zf, zf}, {zf, zf}};

  for (int k0 = 0; k0 < Dq; k0 += 32) {
#pragma unroll
    for (int p = 0; p < 4; ++p) {
      int lin = (p * 128 + tid) * 4;
      int r = lin >> 5, c = lin & 31;
      const __bf16* ag = ctxb + (size_t)(m0 + r) * Dq + k0 + c;
      const float*  wg = wo + (size_t)(n0 + r) * Dq + k0 + c;
      v4bf a4 = *(const v4bf*)ag;
      float4 bv = *(const float4*)wg;
      __builtin_prefetch(ag + 32, 0, 1);
      __builtin_prefetch(wg + 32, 0, 1);
      v4bf b4; b4[0]=(__bf16)bv.x; b4[1]=(__bf16)bv.y; b4[2]=(__bf16)bv.z; b4[3]=(__bf16)bv.w;
      *(v4bf*)&As[r * LDT + c] = a4;
      *(v4bf*)&Bs[r * LDT + c] = b4;
    }
    __syncthreads();
    v16bf af[2], bfr[2];
#pragma unroll
    for (int mt = 0; mt < 2; ++mt) {
      const __bf16* p = &As[(wm * 32 + mt * 16 + lh) * LDT + kh * 8];
      af[mt] = cat16(*(const v8bf*)p, *(const v8bf*)(p + 16));
    }
#pragma unroll
    for (int nt = 0; nt < 2; ++nt) {
      const __bf16* p = &Bs[(wn * 32 + nt * 16 + lh) * LDT + kh * 8];
      bfr[nt] = cat16(*(const v8bf*)p, *(const v8bf*)(p + 16));
    }
#pragma unroll
    for (int mt = 0; mt < 2; ++mt)
#pragma unroll
      for (int nt = 0; nt < 2; ++nt)
        acc[mt][nt] = wmma_bf16(af[mt], bfr[nt], acc[mt][nt]);
    __syncthreads();
  }

#pragma unroll
  for (int mt = 0; mt < 2; ++mt)
#pragma unroll
    for (int nt = 0; nt < 2; ++nt)
#pragma unroll
      for (int r = 0; r < 8; ++r) {
        int m = m0 + wm * 32 + mt * 16 + r + kh * 8;
        int n = n0 + wn * 32 + nt * 16 + lh;
        out[(size_t)m * Dq + n] = acc[mt][nt][r] + bo[n];
      }
}

// ---------------------------------------------------------------------------
extern "C" void kernel_launch(void* const* d_in, const int* in_sizes, int n_in,
                              void* d_out, int out_size, void* d_ws, size_t ws_size,
                              hipStream_t stream) {
  const float* x  = (const float*)d_in[0];
  const float* wq = (const float*)d_in[1];
  const float* wk = (const float*)d_in[2];
  const float* wv = (const float*)d_in[3];
  const float* wo = (const float*)d_in[4];
  const float* bo = (const float*)d_in[5];
  float* out = (float*)d_out;

  const size_t NQ = (size_t)Bq * Hq * Sq * HDq;     // 6,291,456 elems
  __bf16* qo  = (__bf16*)d_ws;                      // [b,h,s,hd] bf16
  __bf16* ko  = qo + NQ;                            // [b,h,s,hd] bf16
  __bf16* vo  = ko + NQ;                            // [b,h,hd,s] bf16 (transposed)
  __bf16* ctx = vo + NQ;                            // [b,s,h*hd] bf16

  dim3 g1(Bq * Sq / 64, Dq / 64, 3), blk(128);
  mha_qkv_gemm<<<g1, blk, 0, stream>>>(x, wq, wk, wv, qo, ko, vo);

  dim3 g2(Sq / 128, Bq * Hq);
  mha_flash_attn<<<g2, blk, 0, stream>>>(qo, ko, vo, ctx);

  dim3 g3(Bq * Sq / 64, Dq / 64);
  mha_out_gemm<<<g3, blk, 0, stream>>>(ctx, wo, bo, out);
}
